// Linear_16733192585553
// MI455X (gfx1250) — matrix operand order repaired and checked by reading, not compiled, not run on hardware
//
#include <hip/hip_runtime.h>
#include <hip/hip_bf16.h>

#define N_TOKENS 16384
#define D_IN     4096
#define D_OUT    4096
#define RANK     16
#define SEG_TOK  2048

#define BM 128
#define BN 128
#define BK 32

typedef __attribute__((ext_vector_type(16))) __bf16 v16bf;
typedef __attribute__((ext_vector_type(8)))  float  v8f;
typedef __attribute__((ext_vector_type(2)))  __bf16 bf16x2;
typedef __attribute__((ext_vector_type(2)))  float  f32x2;

union FragU { unsigned u[8]; v16bf v; };

// f32 pair -> packed bf16 (RNE); should lower to v_cvt_pk_bf16_f32
__device__ __forceinline__ unsigned packcvt(float lo, float hi) {
  f32x2 f = {lo, hi};
  bf16x2 b = __builtin_convertvector(f, bf16x2);
  union { bf16x2 b; unsigned u; } cv;
  cv.b = b;
  return cv.u;
}

// ---------------------------------------------------------------------------
// Kernel 1: H[token][rank] = (x . A_seg[rank]) * per-segment/rank scale
//   seg < 4 (LoRA): scale = lora_scaling[seg]
//   seg >= 4 (VeRA): scale = vera_scaling[g] * vera_d[g][rank]
// ---------------------------------------------------------------------------
__global__ __launch_bounds__(256) void lowrank_h_kernel(
    const float* __restrict__ data,
    const float* __restrict__ lora_a,
    const float* __restrict__ lora_scaling,
    const float* __restrict__ vera_a,
    const float* __restrict__ vera_d,
    const float* __restrict__ vera_scaling,
    float* __restrict__ H)
{
  __shared__ float sX[16][128];
  __shared__ float sAa[16][128];

  const int tok0 = blockIdx.x * 16;
  const int seg  = tok0 >> 11;                       // /2048
  const float* Aptr = (seg < 4) ? (lora_a + (size_t)seg * RANK * D_IN)
                                : (vera_a + (size_t)(seg - 4) * RANK * D_IN);
  const int t    = threadIdx.x;
  const int tok  = t >> 4;
  const int rank = t & 15;

  float acc = 0.f;
  for (int k0 = 0; k0 < D_IN; k0 += 128) {
    #pragma unroll
    for (int i = 0; i < 2; ++i) {                    // 512 float4 / 256 thr
      int idx = t + 256 * i;                         // 0..511
      int row = idx >> 5;                            // 32 float4 per row
      int c   = (idx & 31) << 2;
      *(float4*)&sX [row][c] = *(const float4*)(data + (size_t)(tok0 + row) * D_IN + k0 + c);
      *(float4*)&sAa[row][c] = *(const float4*)(Aptr + (size_t)row        * D_IN + k0 + c);
    }
    __syncthreads();
    #pragma unroll 8
    for (int k = 0; k < 128; ++k)
      acc = fmaf(sX[tok][k], sAa[rank][k], acc);
    __syncthreads();
  }

  float scale;
  if (seg < 4) scale = lora_scaling[seg];
  else         scale = vera_scaling[seg - 4] * vera_d[(seg - 4) * RANK + rank];
  H[(size_t)(tok0 + tok) * RANK + rank] = acc * scale;
}

// ---------------------------------------------------------------------------
// One WMMA K-step (BK=32): 2x4 grid of v_wmma_f32_16x16x32_bf16 per wave.
//   A (16x32): lanes 0-15 row M=lane; VGPR p<4 -> K=2p..2p+1 (+8 if lane>=16),
//              VGPR p>=4 -> K=16+2(p-4).. (+8 if lane>=16)
//   B (32x16): col N=lane%16, same K mapping as A
// ---------------------------------------------------------------------------
__device__ __forceinline__ void wmma_step(const unsigned* __restrict__ uA,
                                          const unsigned* __restrict__ uB,
                                          int wm, int wn, int l16, int half,
                                          v8f (&acc)[2][4])
{
  FragU af[2];
  FragU bfr[4];
  #pragma unroll
  for (int r = 0; r < 2; ++r) {
    int m = wm + r * 16 + l16;
    #pragma unroll
    for (int p = 0; p < 8; ++p) {
      int k = ((p < 4) ? (2 * p) : (16 + 2 * (p - 4))) + half * 8;
      af[r].u[p] = uA[m * (BK / 2) + (k >> 1)];
    }
  }
  #pragma unroll
  for (int c = 0; c < 4; ++c) {
    int n = wn + c * 16 + l16;
    #pragma unroll
    for (int p = 0; p < 8; ++p) {
      int k = ((p < 4) ? (2 * p) : (16 + 2 * (p - 4))) + half * 8;
      bfr[c].u[p] = uB[n * (BK / 2) + (k >> 1)];
    }
  }
  #pragma unroll
  for (int r = 0; r < 2; ++r)
    #pragma unroll
    for (int c = 0; c < 4; ++c)
      acc[r][c] = __builtin_amdgcn_wmma_f32_16x16x32_bf16(
          false, af[r].v, false, bfr[c].v, (short)0, acc[r][c], false, false);
}

// ---------------------------------------------------------------------------
// Kernel 2: fused  out = data @ W^T + bias + (H @ Badj^T)
// 128x128 tile / 256 threads (8 waves, 4x2 wave grid of 32x64 sub-tiles).
// Software-pipelined: double-buffered LDS, one barrier per K-step, global
// loads for tile k+1 in flight across the WMMAs of tile k.
// ---------------------------------------------------------------------------
__global__ __launch_bounds__(256) void fused_lora_gemm_kernel(
    const float* __restrict__ data,
    const float* __restrict__ W,
    const float* __restrict__ bias,
    const float* __restrict__ lora_b,
    const float* __restrict__ vera_b,
    const float* __restrict__ vera_bvec,
    const float* __restrict__ H,
    float* __restrict__ out)
{
  __shared__ unsigned short sA[2][BM * BK];   // bf16 tokens x K (double buffer)
  __shared__ unsigned short sB[2][BN * BK];   // bf16 out-cols x K

  const int t    = threadIdx.x;
  const int n0   = blockIdx.x * BN;
  const int m0   = blockIdx.y * BM;
  const int wave = t >> 5;
  const int lane = t & 31;
  const int half = lane >> 4;
  const int l16  = lane & 15;
  const int wm   = (wave >> 1) * 32;          // wave row base (4 wave-rows)
  const int wn   = (wave & 1) * 64;           // wave col base (2 wave-cols)

  // this thread's fixed staging slots: 4 float4 per matrix per K-step
  int srow[4], scol[4];
  #pragma unroll
  for (int i = 0; i < 4; ++i) {
    int idx = t + 256 * i;                    // 0..1023
    srow[i] = idx >> 3;                       // 8 float4 per 32-wide row
    scol[i] = (idx & 7) << 2;
  }

  v8f acc[2][4];
  #pragma unroll
  for (int r = 0; r < 2; ++r)
    #pragma unroll
    for (int c = 0; c < 4; ++c)
      acc[r][c] = (v8f){0.f, 0.f, 0.f, 0.f, 0.f, 0.f, 0.f, 0.f};

  // prologue: load tile 0 into registers
  float4 ra[4], rb[4];
  #pragma unroll
  for (int i = 0; i < 4; ++i) {
    ra[i] = *(const float4*)(data + (size_t)(m0 + srow[i]) * D_IN + scol[i]);
    rb[i] = *(const float4*)(W    + (size_t)(n0 + srow[i]) * D_IN + scol[i]);
  }

  int p = 0;
  for (int k0 = 0; k0 < D_IN; k0 += BK) {
    // stage tile k0 (registers -> bf16 LDS)
    #pragma unroll
    for (int i = 0; i < 4; ++i) {
      unsigned* da = (unsigned*)&sA[p][srow[i] * BK + scol[i]];
      unsigned* db = (unsigned*)&sB[p][srow[i] * BK + scol[i]];
      da[0] = packcvt(ra[i].x, ra[i].y);
      da[1] = packcvt(ra[i].z, ra[i].w);
      db[0] = packcvt(rb[i].x, rb[i].y);
      db[1] = packcvt(rb[i].z, rb[i].w);
    }
    // issue global loads for tile k0+BK (in flight across the WMMAs below)
    if (k0 + BK < D_IN) {
      #pragma unroll
      for (int i = 0; i < 4; ++i) {
        const float* pa = data + (size_t)(m0 + srow[i]) * D_IN + k0 + BK + scol[i];
        const float* pb = W    + (size_t)(n0 + srow[i]) * D_IN + k0 + BK + scol[i];
        ra[i] = *(const float4*)pa;
        rb[i] = *(const float4*)pb;
        if (i == 0 && k0 + 2 * BK < D_IN) {   // deeper L2 hint
          __builtin_prefetch(pa + BK, 0, 1);
          __builtin_prefetch(pb + BK, 0, 1);
        }
      }
    }
    __syncthreads();                          // tile k0 staged, prior reads done
    wmma_step((const unsigned*)sA[p], (const unsigned*)sB[p],
              wm, wn, l16, half, acc);
    p ^= 1;
  }

  // ---- fused adapter epilogue: one more WMMA round on (H, Badj) ----
  // buffer sA[p]/sB[p] was last read >=1 barrier ago -> safe to overwrite
  const int seg = m0 >> 11;                                  // token segment
  const float* Bsrc = (seg < 4) ? (lora_b + (size_t)seg * D_OUT * RANK)
                                : (vera_b + (size_t)(seg - 4) * D_OUT * RANK);
  const float* bvec = (seg < 4) ? (const float*)0
                                : (vera_bvec + (size_t)(seg - 4) * D_OUT);
  #pragma unroll
  for (int i = 0; i < 4; ++i) {
    int row = srow[i];
    int c   = scol[i];                                       // K base (mult of 4)
    float h0 = 0.f, h1 = 0.f, h2 = 0.f, h3 = 0.f;
    float b0 = 0.f, b1 = 0.f, b2 = 0.f, b3 = 0.f;
    if (c < RANK) {
      const float* hp = H + (size_t)(m0 + row) * RANK + c;
      h0 = hp[0]; h1 = hp[1]; h2 = hp[2]; h3 = hp[3];
      const float* bp = Bsrc + (size_t)(n0 + row) * RANK + c;
      float s = bvec ? bvec[n0 + row] : 1.0f;                // fold vera_bvec
      b0 = bp[0] * s; b1 = bp[1] * s; b2 = bp[2] * s; b3 = bp[3] * s;
    }
    unsigned* da = (unsigned*)&sA[p][row * BK + c];
    unsigned* db = (unsigned*)&sB[p][row * BK + c];
    da[0] = packcvt(h0, h1);
    da[1] = packcvt(h2, h3);
    db[0] = packcvt(b0, b1);
    db[1] = packcvt(b2, b3);
  }
  __syncthreads();
  wmma_step((const unsigned*)sA[p], (const unsigned*)sB[p],
            wm, wn, l16, half, acc);

  // ---- bias + store (C/D layout: VGPR p -> M = p + 8*half, N = l16) ----
  #pragma unroll
  for (int r = 0; r < 2; ++r)
    #pragma unroll
    for (int c = 0; c < 4; ++c) {
      int n = n0 + wn + c * 16 + l16;
      float bs = bias[n];
      #pragma unroll
      for (int q = 0; q < 8; ++q) {
        int m = m0 + wm + r * 16 + half * 8 + q;
        out[(size_t)m * D_OUT + n] = acc[r][c][q] + bs;
      }
    }
}

// ---------------------------------------------------------------------------
extern "C" void kernel_launch(void* const* d_in, const int* in_sizes, int n_in,
                              void* d_out, int out_size, void* d_ws, size_t ws_size,
                              hipStream_t stream) {
  const float* data         = (const float*)d_in[0];
  const float* W            = (const float*)d_in[1];
  const float* bias         = (const float*)d_in[2];
  const float* lora_a       = (const float*)d_in[3];
  const float* lora_b       = (const float*)d_in[4];
  const float* lora_scaling = (const float*)d_in[5];
  const float* vera_a       = (const float*)d_in[6];
  const float* vera_b       = (const float*)d_in[7];
  const float* vera_d       = (const float*)d_in[8];
  const float* vera_bvec    = (const float*)d_in[9];
  const float* vera_scaling = (const float*)d_in[10];
  float*       out          = (float*)d_out;
  float*       H            = (float*)d_ws;        // 16384*16 f32 = 1 MB

  (void)in_sizes; (void)n_in; (void)out_size; (void)ws_size;

  lowrank_h_kernel<<<N_TOKENS / 16, 256, 0, stream>>>(
      data, lora_a, lora_scaling, vera_a, vera_d, vera_scaling, H);

  dim3 grid(D_OUT / BN, N_TOKENS / BM);   // 32 x 128
  fused_lora_gemm_kernel<<<grid, 256, 0, stream>>>(
      data, W, bias, lora_b, vera_b, vera_bvec, H, out);
}
